// MiniMaxM2MoE_6579889898121
// MI455X (gfx1250) — matrix operand order repaired and checked by reading, not compiled, not run on hardware
//
#include <hip/hip_runtime.h>
#include <cstdint>

#define T_TOK 4096
#define H_DIM 1024
#define F_DIM 512
#define E_NUM 16
#define K_TOP 4
#define TM    32   // token rows per block (2 WMMA M-tiles -> 2x B-fragment reuse)

#if __has_builtin(__builtin_amdgcn_global_load_async_to_lds_b128) && \
    __has_builtin(__builtin_amdgcn_s_wait_asynccnt)
#define HAVE_ASYNC_LDS 1
#else
#define HAVE_ASYNC_LDS 0
#endif

typedef __attribute__((ext_vector_type(16))) __bf16 v16bf;
typedef __attribute__((ext_vector_type(8)))  __bf16 v8bf;
typedef __attribute__((ext_vector_type(8)))  float  v8f;
typedef int v4i __attribute__((vector_size(16)));   // matches builtin param type

union AFrag { v16bf v; v8bf h[2]; };

__device__ __forceinline__ unsigned short f2bf(float f) {
  unsigned int u = __float_as_uint(f);
  u = (u + 0x7FFFu + ((u >> 16) & 1u)) >> 16;   // round-to-nearest-even
  return (unsigned short)u;
}

__device__ __forceinline__ v8f wmma_bf16(const AFrag& a, v16bf b, v8f c) {
  return __builtin_amdgcn_wmma_f32_16x16x32_bf16(false, a.v, false, b,
                                                 (short)0, c, false, false);
}

// ---------------------------------------------------------------------------
// fp32 -> bf16 conversion (x and all expert weights)
// ---------------------------------------------------------------------------
__global__ void k_cvt_bf16(const float* __restrict__ in,
                           unsigned short* __restrict__ out, int n) {
  int i = blockIdx.x * blockDim.x + threadIdx.x;
  if (i < n) out[i] = f2bf(in[i]);
}

// ---------------------------------------------------------------------------
// Router: one wave32 per token. fp32 logits -> sigmoid -> top-4 of
// (score + bias) -> renormalized raw scores into dense combine[T,E].
// ---------------------------------------------------------------------------
__global__ __launch_bounds__(256)
void k_router(const float* __restrict__ x, const float* __restrict__ gw,
              const float* __restrict__ bias, float* __restrict__ combine) {
  const int t    = blockIdx.x * 8 + (threadIdx.x >> 5);
  const int lane = threadIdx.x & 31;

  float acc[E_NUM];
#pragma unroll
  for (int e = 0; e < E_NUM; ++e) acc[e] = 0.f;

  for (int i = lane; i < H_DIM; i += 32) {
    float xv = x[(size_t)t * H_DIM + i];
#pragma unroll
    for (int e = 0; e < E_NUM; ++e) acc[e] += xv * gw[e * H_DIM + i];
  }
#pragma unroll
  for (int m = 16; m >= 1; m >>= 1) {
#pragma unroll
    for (int e = 0; e < E_NUM; ++e) acc[e] += __shfl_xor(acc[e], m, 32);
  }

  float s[E_NUM];
#pragma unroll
  for (int e = 0; e < E_NUM; ++e) s[e] = 1.f / (1.f + __expf(-acc[e]));

  bool sel[E_NUM];
#pragma unroll
  for (int e = 0; e < E_NUM; ++e) sel[e] = false;
  float wsum = 0.f;
  for (int k = 0; k < K_TOP; ++k) {          // strict > keeps lowest index on tie
    float best = -1e30f; int bi = 0;
    for (int e = 0; e < E_NUM; ++e) {
      if (!sel[e]) { float v = s[e] + bias[e]; if (v > best) { best = v; bi = e; } }
    }
    sel[bi] = true; wsum += s[bi];
  }
  if (lane < E_NUM)
    combine[(size_t)t * E_NUM + lane] = sel[lane] ? (s[lane] / wsum) : 0.f;
}

// ---------------------------------------------------------------------------
// Fused expert kernel: 32-token tile per block, dense loop over 16 experts.
// bf16 WMMA / fp32 accumulate; output tile lives in VGPRs across the expert
// loop (no atomics -> deterministic). x tile staged via async loads to LDS.
// ---------------------------------------------------------------------------
__global__ __launch_bounds__(256)
void k_moe(const unsigned short* __restrict__ xb,
           const unsigned short* __restrict__ w1b,
           const unsigned short* __restrict__ w3b,
           const unsigned short* __restrict__ w2b,
           const float* __restrict__ combine,
           float* __restrict__ out) {
  __shared__ __align__(16) unsigned short lds_x[TM * H_DIM];   // 64 KB
  __shared__ __align__(16) unsigned short lds_h[TM * F_DIM];   // 32 KB
  __shared__ float lds_c[TM];

  const int tid  = threadIdx.x;
  const int wave = tid >> 5;
  const int lane = tid & 31;
  const int m0   = blockIdx.x * TM;
  const int r16  = lane & 15;
  const int hsel = lane >> 4;       // 0: K lo-half group, 1: K hi-half group
  const v8f zero = {};

  // ---- stage x tile into LDS (CDNA5 async-to-LDS path when available) ----
  {
    const unsigned short* gsrc = xb + (size_t)m0 * H_DIM;
#if HAVE_ASYNC_LDS
    for (int i = tid * 8; i < TM * H_DIM; i += 256 * 8) {
      __builtin_amdgcn_global_load_async_to_lds_b128(
          (__attribute__((address_space(1))) v4i*)(gsrc + i),
          (__attribute__((address_space(3))) v4i*)(&lds_x[i]),
          0, 0);
    }
    __builtin_amdgcn_s_wait_asynccnt(0);
#else
    for (int i = tid; i < TM * H_DIM; i += 256) lds_x[i] = gsrc[i];
#endif
  }
  __syncthreads();

  // persistent output accumulators: [mi(2)][j(8)] 16x16 tiles
  v8f oacc[16];
#pragma unroll
  for (int i = 0; i < 16; ++i) oacc[i] = zero;

  for (int e = 0; e < E_NUM; ++e) {
    if (tid < TM) lds_c[tid] = combine[(size_t)(m0 + tid) * E_NUM + e];

    const unsigned short* w1e = w1b + (size_t)e * F_DIM * H_DIM;
    const unsigned short* w3e = w3b + (size_t)e * F_DIM * H_DIM;
    const unsigned short* w2e = w2b + (size_t)e * H_DIM * F_DIM;
    __builtin_prefetch(w1e, 0, 1);
    __builtin_prefetch(w3e, 0, 1);
    __builtin_prefetch(w2e, 0, 1);

    // ---- GEMM1: g = x@W1^T, u = x@W3^T ; per wave 4 f-tiles x 2 m-tiles ----
    v8f accg[8], accu[8];               // [j(4)][mi(2)]
#pragma unroll
    for (int i = 0; i < 8; ++i) { accg[i] = zero; accu[i] = zero; }

    for (int k0 = 0; k0 < H_DIM; k0 += 32) {
      AFrag a0, a1;
      const unsigned short* x0 = &lds_x[r16 * H_DIM + k0 + hsel * 16];
      const unsigned short* x1 = &lds_x[(16 + r16) * H_DIM + k0 + hsel * 16];
      a0.h[0] = *(const v8bf*)(x0); a0.h[1] = *(const v8bf*)(x0 + 8);
      a1.h[0] = *(const v8bf*)(x1); a1.h[1] = *(const v8bf*)(x1 + 8);
#pragma unroll
      for (int j = 0; j < 4; ++j) {
        const int n0 = (wave * 4 + j) * 16;
        const size_t boff = (size_t)(n0 + r16) * H_DIM + k0 + hsel * 16;
        v16bf bg = *(const v16bf*)&w1e[boff];
        v16bf bu = *(const v16bf*)&w3e[boff];
        accg[j * 2 + 0] = wmma_bf16(a0, bg, accg[j * 2 + 0]);
        accg[j * 2 + 1] = wmma_bf16(a1, bg, accg[j * 2 + 1]);
        accu[j * 2 + 0] = wmma_bf16(a0, bu, accu[j * 2 + 0]);
        accu[j * 2 + 1] = wmma_bf16(a1, bu, accu[j * 2 + 1]);
      }
    }

    // silu(g) * u  -> bf16 h tile in LDS (C/D layout: vgpr r = row r / r+8)
#pragma unroll
    for (int j = 0; j < 4; ++j) {
      const int col = (wave * 4 + j) * 16 + r16;
#pragma unroll
      for (int mi = 0; mi < 2; ++mi) {
#pragma unroll
        for (int r = 0; r < 8; ++r) {
          const int row = mi * 16 + (hsel ? (r + 8) : r);
          float g  = accg[j * 2 + mi][r];
          float u  = accu[j * 2 + mi][r];
          float hv = (g / (1.f + __expf(-g))) * u;
          lds_h[row * F_DIM + col] = f2bf(hv);
        }
      }
    }
    __syncthreads();

    // ---- GEMM2: y = h@W2^T ; per wave 8 n-tiles x 2 m-tiles; oacc += c*y ----
#pragma unroll
    for (int j = 0; j < 8; ++j) {
      const int n0 = (wave * 8 + j) * 16;
      v8f acc0 = zero, acc1 = zero;
      for (int k0 = 0; k0 < F_DIM; k0 += 32) {
        AFrag a0, a1;
        const unsigned short* h0 = &lds_h[r16 * F_DIM + k0 + hsel * 16];
        const unsigned short* h1 = &lds_h[(16 + r16) * F_DIM + k0 + hsel * 16];
        a0.h[0] = *(const v8bf*)(h0); a0.h[1] = *(const v8bf*)(h0 + 8);
        a1.h[0] = *(const v8bf*)(h1); a1.h[1] = *(const v8bf*)(h1 + 8);
        v16bf bw = *(const v16bf*)&w2e[(size_t)(n0 + r16) * F_DIM + k0 + hsel * 16];
        acc0 = wmma_bf16(a0, bw, acc0);
        acc1 = wmma_bf16(a1, bw, acc1);
      }
#pragma unroll
      for (int r = 0; r < 8; ++r) {
        const int row = hsel ? (r + 8) : r;
        oacc[j][r]     += lds_c[row]      * acc0[r];
        oacc[8 + j][r] += lds_c[16 + row] * acc1[r];
      }
    }
    __syncthreads();   // protect lds_h / lds_c before next expert
  }

  // ---- write final fp32 output tile from register accumulators ----
#pragma unroll
  for (int j = 0; j < 8; ++j) {
    const int col = (wave * 8 + j) * 16 + r16;
#pragma unroll
    for (int mi = 0; mi < 2; ++mi) {
#pragma unroll
      for (int r = 0; r < 8; ++r) {
        const int row = mi * 16 + (hsel ? (r + 8) : r);
        out[(size_t)(m0 + row) * H_DIM + col] = oacc[mi * 8 + j][r];
      }
    }
  }
}

// ---------------------------------------------------------------------------
extern "C" void kernel_launch(void* const* d_in, const int* in_sizes, int n_in,
                              void* d_out, int out_size, void* d_ws, size_t ws_size,
                              hipStream_t stream) {
  (void)in_sizes; (void)n_in; (void)out_size; (void)ws_size;

  const float* x    = (const float*)d_in[0];
  const float* gw   = (const float*)d_in[1];
  const float* bias = (const float*)d_in[2];
  const float* w1   = (const float*)d_in[3];
  const float* w3   = (const float*)d_in[4];
  const float* w2   = (const float*)d_in[5];
  float* out = (float*)d_out;

  // Workspace layout (bf16 copies + combine), ~56.6 MB total.
  unsigned short* xb  = (unsigned short*)d_ws;
  unsigned short* w1b = xb  + (size_t)T_TOK * H_DIM;
  unsigned short* w3b = w1b + (size_t)E_NUM * F_DIM * H_DIM;
  unsigned short* w2b = w3b + (size_t)E_NUM * F_DIM * H_DIM;
  float* combine = (float*)(w2b + (size_t)E_NUM * H_DIM * F_DIM);

  const int nX = T_TOK * H_DIM;
  const int nW = E_NUM * F_DIM * H_DIM;
  k_cvt_bf16<<<(nX + 255) / 256, 256, 0, stream>>>(x,  xb,  nX);
  k_cvt_bf16<<<(nW + 255) / 256, 256, 0, stream>>>(w1, w1b, nW);
  k_cvt_bf16<<<(nW + 255) / 256, 256, 0, stream>>>(w3, w3b, nW);
  k_cvt_bf16<<<(nW + 255) / 256, 256, 0, stream>>>(w2, w2b, nW);

  k_router<<<T_TOK / 8, 256, 0, stream>>>(x, gw, bias, combine);

  k_moe<<<T_TOK / TM, 256, 0, stream>>>(xb, w1b, w3b, w2b, combine, out);
}
